// EfficientFrequencyAttention_46986942218810
// MI455X (gfx1250) — matrix operand order repaired and checked by reading, not compiled, not run on hardware
//
#include <hip/hip_runtime.h>

#define DEV __device__ __forceinline__

typedef __attribute__((ext_vector_type(16))) __bf16 bf16x16;
typedef __attribute__((ext_vector_type(8)))  __bf16 bf16x8;
typedef __attribute__((ext_vector_type(8)))  float  f32x8;

// ---------------------------------------------------------------- constants
// N=8, C=128, H=W=128
static constexpr int kHW   = 16384;      // 128*128
static constexpr int kTot  = 16777216;   // 8*128*16384

// ---------------------------------------------------------------- helpers
DEV f32x8 zero8() { f32x8 z = {0.f,0.f,0.f,0.f,0.f,0.f,0.f,0.f}; return z; }

union FragU { bf16x16 v; bf16x8 h[2]; };

// A fragment (16x32 MxK, bf16): lane<16 -> K 0..7 & 16..23 ; lane>=16 -> K 8..15 & 24..31
DEV bf16x16 frag_a(const __bf16* rowk, int lane) {
  const int klo = (lane >> 4) * 8;
  FragU f;
  f.h[0] = *reinterpret_cast<const bf16x8*>(rowk + klo);
  f.h[1] = *reinterpret_cast<const bf16x8*>(rowk + klo + 16);
  return f.v;
}
// B fragment (32x16 KxN, bf16): lane<16 -> K 0..7 & 16..23 of column n ; lane>=16 -> K 8..15 & 24..31
DEV bf16x16 frag_b(const __bf16* colk, int lane) {
  const int klo = (lane >> 4) * 8;
  FragU f;
  f.h[0] = *reinterpret_cast<const bf16x8*>(colk + klo);
  f.h[1] = *reinterpret_cast<const bf16x8*>(colk + klo + 16);
  return f.v;
}
DEV f32x8 wmma_bf16(bf16x16 a, bf16x16 b, f32x8 c) {
  return __builtin_amdgcn_wmma_f32_16x16x32_bf16(false, a, false, b, (short)0, c, false, false);
}
DEV float gelu_f(float v) { return 0.5f * v * (1.0f + erff(v * 0.70710678118654752f)); }

DEV float block_reduce_max(float v, float* red, int tid) {
  red[tid] = v; __syncthreads();
  for (int s = 128; s > 0; s >>= 1) { if (tid < s) red[tid] = fmaxf(red[tid], red[tid + s]); __syncthreads(); }
  float r = red[0]; __syncthreads(); return r;
}
DEV float block_reduce_sum(float v, float* red, int tid) {
  red[tid] = v; __syncthreads();
  for (int s = 128; s > 0; s >>= 1) { if (tid < s) red[tid] += red[tid + s]; __syncthreads(); }
  float r = red[0]; __syncthreads(); return r;
}

// ---------------------------------------------------------------- utility kernels
__global__ void cvt_bf16_k(const float* __restrict__ a, __bf16* __restrict__ o, int n) {
  for (int i = blockIdx.x * blockDim.x + threadIdx.x; i < n; i += gridDim.x * blockDim.x)
    o[i] = (__bf16)a[i];
}
__global__ void zerof_k(float* __restrict__ p, int n) {
  for (int i = blockIdx.x * blockDim.x + threadIdx.x; i < n; i += gridDim.x * blockDim.x)
    p[i] = 0.0f;
}
// OIHW 3x3 [oc][ic][t]  ->  tap-major bf16 [t][oc][ic]
__global__ void reorder3_k(const float* __restrict__ w, __bf16* __restrict__ o) {
  int i = blockIdx.x * 256 + threadIdx.x;          // grid covers exactly 9*128*128
  int t = i >> 14, rem = i & 16383, oc = rem >> 7, ic = rem & 127;
  o[i] = (__bf16)w[((size_t)oc * 128 + ic) * 9 + t];
}

// ---------------------------------------------------------------- depthwise 3x3 + gelu (x1 -> B_re ch 0..63)
__global__ __launch_bounds__(256)
void dwconv_gelu_k(const float* __restrict__ x, const float* __restrict__ w,
                   const float* __restrict__ bias, __bf16* __restrict__ Bre) {
  __shared__ float wk[9];
  const int j = blockIdx.x & 63, n = blockIdx.x >> 6;
  if (threadIdx.x < 9) wk[threadIdx.x] = w[j * 9 + threadIdx.x];
  __syncthreads();
  const size_t bin  = ((size_t)n * 128 + 2 * j) << 14;
  const size_t bout = ((size_t)n * 128 + j) << 14;
  const float bj = bias[j];
  for (int p = threadIdx.x; p < kHW; p += 256) {
    const int y = p >> 7, xx = p & 127;
    float acc = bj;
    for (int ky = 0; ky < 3; ++ky)
      for (int kx = 0; kx < 3; ++kx) {
        const int yy = y + ky - 1, x2 = xx + kx - 1;
        if (yy >= 0 && yy < 128 && x2 >= 0 && x2 < 128)
          acc += wk[ky * 3 + kx] * x[bin + (yy << 7) + x2];
      }
    Bre[bout + p] = (__bf16)gelu_f(acc);
  }
}

// ---------------------------------------------------------------- qkvl: 160x64 GEMM + gelu, routed per head
__global__ __launch_bounds__(320)
void qkvl_wmma_k(const __bf16* __restrict__ xb, const __bf16* __restrict__ wq,
                 const float* __restrict__ bias, float* __restrict__ qsum,
                 float* __restrict__ kbuf, float* __restrict__ vbuf, __bf16* __restrict__ Bre) {
  __shared__ __bf16 As[160][64];
  __shared__ __bf16 Bs[32][64];
  const int tid = threadIdx.x, lane = tid & 31, wave = tid >> 5;
  const int p0 = (blockIdx.x & 511) * 32, n = blockIdx.x >> 9;
  // stage weights [160][64]
  for (int ci = tid; ci < 640; ci += 320) {
    const int oc = ci >> 2, part = ci & 3;
    *reinterpret_cast<bf16x16*>(&As[oc][part * 16]) =
        *reinterpret_cast<const bf16x16*>(wq + (size_t)oc * 64 + part * 16);
  }
  // stage B: Bs[nn][kk] = xb[n][2*kk+1][p0+nn]
  for (int idx = tid; idx < 2048; idx += 320) {
    const int kk = idx >> 5, nn = idx & 31;
    Bs[nn][kk] = xb[(((size_t)n * 128 + 2 * kk + 1) << 14) + p0 + nn];
  }
  __syncthreads();
  const int oc0 = wave * 16;
  f32x8 acc0 = zero8(), acc1 = zero8();
  #pragma unroll
  for (int ks = 0; ks < 2; ++ks) {
    bf16x16 a  = frag_a(&As[oc0 + (lane & 15)][ks * 32], lane);
    bf16x16 b0 = frag_b(&Bs[lane & 15][ks * 32], lane);
    bf16x16 b1 = frag_b(&Bs[16 + (lane & 15)][ks * 32], lane);
    acc0 = wmma_bf16(a, b0, acc0);
    acc1 = wmma_bf16(a, b1, acc1);
  }
  const int col = lane & 15, mhi = (lane >> 4) * 8;
  #pragma unroll
  for (int r = 0; r < 8; ++r) {
    const int oc = oc0 + r + mhi, head = oc >> 5, c = oc & 31;
    const float bz = bias[oc];
    #pragma unroll
    for (int nt = 0; nt < 2; ++nt) {
      const float g = gelu_f((nt ? acc1[r] : acc0[r]) + bz);
      const int p = p0 + nt * 16 + col;
      const size_t b32 = (((size_t)n * 32 + c) << 14) + p;
      if (head <= 1)      atomicAdd(&qsum[b32], g);
      else if (head == 2) kbuf[b32] = g;
      else if (head == 3) vbuf[b32] = g;
      else Bre[(((size_t)n * 128 + 64 + c) << 14) + p] = (__bf16)g;
    }
  }
}

// ---------------------------------------------------------------- pooling (avg3x3 s2 SAME, max2x2 s2)
__global__ __launch_bounds__(256)
void pool_qk_k(const float* __restrict__ qs, const float* __restrict__ kb,
               float* __restrict__ qp, float* __restrict__ kp) {
  const size_t bin = (size_t)blockIdx.x << 14, bo = (size_t)blockIdx.x << 12;
  for (int o = threadIdx.x; o < 4096; o += 256) {
    const int oy = o >> 6, ox = o & 63;
    float s = 0.0f;
    for (int dy = -1; dy <= 1; ++dy)
      for (int dx = -1; dx <= 1; ++dx) {
        const int iy = 2 * oy + dy, ix = 2 * ox + dx;
        if (iy >= 0 && iy < 128 && ix >= 0 && ix < 128) s += qs[bin + (iy << 7) + ix];
      }
    qp[bo + o] = s * (1.0f / 9.0f);
    float m = -3.0e38f;
    for (int dy = 0; dy < 2; ++dy)
      for (int dx = 0; dx < 2; ++dx)
        m = fmaxf(m, kb[bin + ((2 * oy + dy) << 7) + 2 * ox + dx]);
    kp[bo + o] = m;
  }
}

// ---------------------------------------------------------------- qk = q@k^T (32x32, K=4096) + column softmax
__global__ __launch_bounds__(256)
void qk_softmax_k(const float* __restrict__ qp, const float* __restrict__ kp, float* __restrict__ A) {
  __shared__ float qk[32][32];
  const int n = blockIdx.x, tid = threadIdx.x;
  for (int e = tid * 4; e < tid * 4 + 4; ++e) {
    const int c = e >> 5, d = e & 31;
    const float* qr = qp + (((size_t)n * 32 + c) << 12);
    const float* kr = kp + (((size_t)n * 32 + d) << 12);
    float s = 0.0f;
    for (int p = 0; p < 4096; ++p) s += qr[p] * kr[p];
    qk[c][d] = s;
  }
  __syncthreads();
  if (tid < 32) {
    const int d = tid;
    float m = -3.0e38f;
    for (int c = 0; c < 32; ++c) m = fmaxf(m, qk[c][d]);
    float z = 0.0f;
    for (int c = 0; c < 32; ++c) z += expf(qk[c][d] - m);
    const float iz = 1.0f / z;
    for (int c = 0; c < 32; ++c) A[(((size_t)n * 32 + c) << 5) + d] = expf(qk[c][d] - m) * iz;
  }
}

// ---------------------------------------------------------------- x2o = A^T @ v  -> B_re ch 96..127 (bf16)
__global__ __launch_bounds__(256)
void x2o_k(const float* __restrict__ A, const float* __restrict__ v, __bf16* __restrict__ Bre) {
  __shared__ float As[32][32];
  const int n = blockIdx.x >> 6, pc = blockIdx.x & 63;
  const int p = pc * 256 + threadIdx.x;
  float* Af = &As[0][0];
  for (int i = threadIdx.x; i < 1024; i += 256) Af[i] = A[(size_t)n * 1024 + i];
  __syncthreads();
  float vv[32];
  #pragma unroll
  for (int q = 0; q < 32; ++q) vv[q] = v[(((size_t)n * 32 + q) << 14) + p];
  #pragma unroll
  for (int k = 0; k < 32; ++k) {
    float s = 0.0f;
    #pragma unroll
    for (int q = 0; q < 32; ++q) s += As[q][k] * vv[q];
    Bre[(((size_t)n * 128 + 96 + k) << 14) + p] = (__bf16)s;
  }
}

// ---------------------------------------------------------------- generic 128ch implicit-GEMM conv (WMMA)
// TAPS: 9 (3x3 SAME) or 1 (1x1).  EPI: 0 raw f32, 1 bias+f32, 2 bias+bf16
template <int TAPS, int EPI>
__global__ __launch_bounds__(256)
void conv128_wmma_k(const __bf16* __restrict__ in, const __bf16* __restrict__ wt,
                    const float* __restrict__ bias, float* __restrict__ outf,
                    __bf16* __restrict__ outb) {
  __shared__ __bf16 As[128][32];
  __shared__ __bf16 Bs[32][32];
  const int tid = threadIdx.x, lane = tid & 31, wave = tid >> 5;
  const int xc = blockIdx.x & 3, y = (blockIdx.x >> 2) & 127, n = blockIdx.x >> 9;
  const int x0 = xc * 32, oc0 = wave * 16;
  f32x8 acc0 = zero8(), acc1 = zero8();
  for (int t = 0; t < TAPS; ++t) {
    const int dy = (TAPS == 9) ? (t / 3 - 1) : 0;
    const int dx = (TAPS == 9) ? (t % 3 - 1) : 0;
    const int yy = y + dy;
    const bool yok = (yy >= 0) && (yy < 128);
    const int yyc = yy < 0 ? 0 : (yy > 127 ? 127 : yy);
    for (int icc = 0; icc < 4; ++icc) {
      const int ic0 = icc * 32;
      { // stage A: weights [128 oc][32 k]
        const int oc = tid >> 1, half = tid & 1;
        *reinterpret_cast<bf16x16*>(&As[oc][half * 16]) =
            *reinterpret_cast<const bf16x16*>(wt + ((size_t)t * 128 + oc) * 128 + ic0 + half * 16);
      }
      { // stage B: Bs[nn][kk] = in[n][ic0+kk][yy][x0+dx+nn]
        const int kk = tid >> 3, nn0 = (tid & 7) * 4;
        const size_t rowbase = (((size_t)n * 128 + ic0 + kk) << 14) + (yyc << 7);
        if (yok) __builtin_prefetch(in + rowbase + x0, 0, 0);  // global_prefetch_b8
        #pragma unroll
        for (int j = 0; j < 4; ++j) {
          const int nn = nn0 + j, xx = x0 + nn + dx;
          __bf16 vv = (__bf16)0.0f;
          if (yok && xx >= 0 && xx < 128) vv = in[rowbase + xx];
          Bs[nn][kk] = vv;
        }
      }
      __syncthreads();
      bf16x16 a  = frag_a(&As[oc0 + (lane & 15)][0], lane);
      bf16x16 b0 = frag_b(&Bs[lane & 15][0], lane);
      bf16x16 b1 = frag_b(&Bs[16 + (lane & 15)][0], lane);
      acc0 = wmma_bf16(a, b0, acc0);
      acc1 = wmma_bf16(a, b1, acc1);
      __syncthreads();
    }
  }
  const int col = lane & 15, mhi = (lane >> 4) * 8;
  #pragma unroll
  for (int r = 0; r < 8; ++r) {
    const int oc = oc0 + r + mhi;
    float v0 = acc0[r], v1 = acc1[r];
    if (EPI >= 1) { const float bz = bias[oc]; v0 += bz; v1 += bz; }
    const size_t base = (((size_t)n * 128 + oc) << 14) + (y << 7) + x0;
    if (EPI == 2) { outb[base + col] = (__bf16)v0; outb[base + 16 + col] = (__bf16)v1; }
    else          { outf[base + col] = v0;         outf[base + 16 + col] = v1; }
  }
}

// ---------------------------------------------------------------- separable gaussians (f32, exact borders)
__global__ __launch_bounds__(256)
void gauss_h_k(const float* __restrict__ in, float* __restrict__ out, int ks, float sigma) {
  float g[7];
  const int r = ks >> 1;
  float sum = 0.0f;
  for (int i = 0; i < ks; ++i) { g[i] = expf(-(float)((i - r) * (i - r)) / (2.0f * sigma * sigma)); sum += g[i]; }
  for (int i = 0; i < ks; ++i) g[i] /= sum;
  const size_t base = (size_t)blockIdx.x << 14;
  for (int p = threadIdx.x; p < kHW; p += 256) {
    const int y = p >> 7, xx = p & 127;
    float acc = 0.0f;
    for (int i = 0; i < ks; ++i) {
      const int x2 = xx + i - r;
      if (x2 >= 0 && x2 < 128) acc += g[i] * in[base + (y << 7) + x2];
    }
    out[base + p] = acc;
  }
}
__global__ __launch_bounds__(256)
void gauss_v_k(const float* __restrict__ in, float* __restrict__ out, int ks, float sigma,
               const float* __restrict__ resid) {
  float g[7];
  const int r = ks >> 1;
  float sum = 0.0f;
  for (int i = 0; i < ks; ++i) { g[i] = expf(-(float)((i - r) * (i - r)) / (2.0f * sigma * sigma)); sum += g[i]; }
  for (int i = 0; i < ks; ++i) g[i] /= sum;
  const size_t base = (size_t)blockIdx.x << 14;
  for (int p = threadIdx.x; p < kHW; p += 256) {
    const int y = p >> 7, xx = p & 127;
    float acc = 0.0f;
    for (int i = 0; i < ks; ++i) {
      const int y2 = y + i - r;
      if (y2 >= 0 && y2 < 128) acc += g[i] * in[base + (y2 << 7) + xx];
    }
    out[base + p] = resid ? (resid[base + p] - acc) : acc;
  }
}

// ---------------------------------------------------------------- softmax stats
// rows of 16384 contiguous (ctx):   stat = {max, 1/sum_exp}
__global__ __launch_bounds__(256)
void rowstats_k(const float* __restrict__ v, float* __restrict__ stat) {
  __shared__ float red[256];
  const size_t base = (size_t)blockIdx.x << 14;
  float m = -3.0e38f;
  for (int i = threadIdx.x; i < kHW; i += 256) m = fmaxf(m, v[base + i]);
  m = block_reduce_max(m, red, threadIdx.x);
  float s = 0.0f;
  for (int i = threadIdx.x; i < kHW; i += 256) s += expf(v[base + i] - m);
  s = block_reduce_sum(s, red, threadIdx.x);
  if (threadIdx.x == 0) { stat[2 * blockIdx.x] = m; stat[2 * blockIdx.x + 1] = 1.0f / s; }
}
// columns of qx: per (n,w), elements stride 128 (16384 of them)
__global__ __launch_bounds__(256)
void colstats_k(const float* __restrict__ qx, float* __restrict__ stat) {
  __shared__ float red[256];
  const int n = blockIdx.x >> 7, w = blockIdx.x & 127;
  const float* base = qx + ((size_t)n << 21) + w;
  float m = -3.0e38f;
  for (int i = threadIdx.x; i < kHW; i += 256) m = fmaxf(m, base[(size_t)i << 7]);
  m = block_reduce_max(m, red, threadIdx.x);
  float s = 0.0f;
  for (int i = threadIdx.x; i < kHW; i += 256) s += expf(base[(size_t)i << 7] - m);
  s = block_reduce_sum(s, red, threadIdx.x);
  if (threadIdx.x == 0) { stat[2 * blockIdx.x] = m; stat[2 * blockIdx.x + 1] = 1.0f / s; }
}

// ---------------------------------------------------------------- att[n] = softmax(ctx) @ ctx^T  (K = 16384)
__global__ __launch_bounds__(256)
void att_wmma_k(const float* __restrict__ ctx, const float* __restrict__ cstat, float* __restrict__ att) {
  __shared__ __bf16 As[128][32];
  __shared__ __bf16 Bs[128][32];
  const int n = blockIdx.x, tid = threadIdx.x, lane = tid & 31, wave = tid >> 5;
  const int c0 = wave * 16;
  f32x8 acc[8];
  #pragma unroll
  for (int j = 0; j < 8; ++j) acc[j] = zero8();
  for (int p0 = 0; p0 < kHW; p0 += 32) {
    const int row = tid >> 1, half = tid & 1;
    const float* src = ctx + (((size_t)n * 128 + row) << 14) + p0 + half * 16;
    const float m = cstat[2 * (n * 128 + row)], iz = cstat[2 * (n * 128 + row) + 1];
    #pragma unroll
    for (int i = 0; i < 16; ++i) {
      const float v = src[i];
      As[row][half * 16 + i] = (__bf16)(expf(v - m) * iz);
      Bs[row][half * 16 + i] = (__bf16)v;
    }
    __syncthreads();
    bf16x16 a = frag_a(&As[c0 + (lane & 15)][0], lane);
    #pragma unroll
    for (int j = 0; j < 8; ++j) {
      bf16x16 b = frag_b(&Bs[j * 16 + (lane & 15)][0], lane);
      acc[j] = wmma_bf16(a, b, acc[j]);
    }
    __syncthreads();
  }
  const int col = lane & 15, mhi = (lane >> 4) * 8;
  #pragma unroll
  for (int j = 0; j < 8; ++j)
    #pragma unroll
    for (int r = 0; r < 8; ++r)
      att[(((size_t)n * 128 + c0 + r + mhi) << 7) + j * 16 + col] = acc[j][r];
}

// ---------------------------------------------------------------- freq = x + att^T @ S  (S built on the fly)
__global__ __launch_bounds__(256)
void freq_wmma_k(const float* __restrict__ att, const float* __restrict__ qx,
                 const float* __restrict__ sstat, const float* __restrict__ x,
                 __bf16* __restrict__ freqb) {
  __shared__ __bf16 As[128][32];
  __shared__ __bf16 Bs[32][32];
  const int tid = threadIdx.x, lane = tid & 31, wave = tid >> 5;
  const int q0 = (blockIdx.x & 511) * 32, n = blockIdx.x >> 9;
  const int cc0 = wave * 16;
  f32x8 acc0 = zero8(), acc1 = zero8();
  for (int j0 = 0; j0 < 128; j0 += 32) {
    { // As[cc][kk] = att[n][j0+kk][cc]
      const int r = tid >> 3, seg = tid & 7;
      const float* src = att + (((size_t)n * 128 + j0 + r) << 7) + seg * 16;
      #pragma unroll
      for (int i = 0; i < 16; ++i) As[seg * 16 + i][r] = (__bf16)src[i];
    }
    { // Bs[nn][kk] = S[n, q0+nn, j0+kk]
      const int nn = tid >> 3, seg = tid & 7, q = q0 + nn;
      const float* src = qx + ((((size_t)n * 128 + (q >> 7)) << 7) + (q & 127)) * 128 + j0 + seg * 4;
      #pragma unroll
      for (int i = 0; i < 4; ++i) {
        const int w = j0 + seg * 4 + i;
        Bs[nn][seg * 4 + i] = (__bf16)(expf(src[i] - sstat[2 * (n * 128 + w)]) * sstat[2 * (n * 128 + w) + 1]);
      }
    }
    __syncthreads();
    bf16x16 a  = frag_a(&As[cc0 + (lane & 15)][0], lane);
    bf16x16 b0 = frag_b(&Bs[lane & 15][0], lane);
    bf16x16 b1 = frag_b(&Bs[16 + (lane & 15)][0], lane);
    acc0 = wmma_bf16(a, b0, acc0);
    acc1 = wmma_bf16(a, b1, acc1);
    __syncthreads();
  }
  const int col = lane & 15, mhi = (lane >> 4) * 8;
  #pragma unroll
  for (int r = 0; r < 8; ++r) {
    const int cc = cc0 + r + mhi;
    const size_t base = (((size_t)n * 128 + cc) << 14) + q0;
    freqb[base + col]      = (__bf16)(x[base + col] + acc0[r]);
    freqb[base + 16 + col] = (__bf16)(x[base + 16 + col] + acc1[r]);
  }
}

// ---------------------------------------------------------------- batch-norm stats -> {scale, shift} per channel
__global__ __launch_bounds__(256)
void bnstats_k(const float* __restrict__ z, const float* __restrict__ g,
               const float* __restrict__ b, float* __restrict__ bnp) {
  __shared__ float red[256];
  const int c = blockIdx.x;
  float s1 = 0.0f, s2 = 0.0f;
  for (int n = 0; n < 8; ++n) {
    const float* base = z + (((size_t)n * 128 + c) << 14);
    for (int i = threadIdx.x; i < kHW; i += 256) { const float a = base[i]; s1 += a; s2 += a * a; }
  }
  const float t1 = block_reduce_sum(s1, red, threadIdx.x);
  const float t2 = block_reduce_sum(s2, red, threadIdx.x);
  if (threadIdx.x == 0) {
    const float mean = t1 * (1.0f / 131072.0f);
    const float var  = t2 * (1.0f / 131072.0f) - mean * mean;
    const float sc   = g[c] * rsqrtf(var + 1e-5f);
    bnp[2 * c] = sc; bnp[2 * c + 1] = b[c] - mean * sc;
  }
}
template <int OB>
__global__ __launch_bounds__(256)
void bnrelu_k(const float* __restrict__ z, const float* __restrict__ bnp,
              float* __restrict__ of, __bf16* __restrict__ ob) {
  for (int i = blockIdx.x * 256 + threadIdx.x; i < kTot; i += gridDim.x * 256) {
    const int c = (i >> 14) & 127;
    const float v = fmaxf(0.0f, z[i] * bnp[2 * c] + bnp[2 * c + 1]);
    if (OB) ob[i] = (__bf16)v; else of[i] = v;
  }
}

// ---------------------------------------------------------------- p-norm, 1D convs, combine
__global__ __launch_bounds__(256)
void pnorm_k(const float* __restrict__ v, float* __restrict__ t, int off) {
  __shared__ float red[256];
  const float* base = v + ((size_t)blockIdx.x << 14);
  float s = 0.0f;
  for (int i = threadIdx.x; i < kHW; i += 256) { const float a = base[i]; s += a * a * a; }
  const float tot = block_reduce_sum(s, red, threadIdx.x);
  if (threadIdx.x == 0) {
    const int n = blockIdx.x >> 7, c = blockIdx.x & 127;
    t[n * 256 + 2 * c + off] = cbrtf(tot * (1.0f / 16384.0f) + 1e-12f);
  }
}
__global__ __launch_bounds__(256)
void dvec_k(const float* __restrict__ t, const float* __restrict__ c1,
            const float* __restrict__ c2, float* __restrict__ d01) {
  __shared__ float u[256];
  const int n = blockIdx.x, i = threadIdx.x;
  float a = 0.0f;
  for (int j = 0; j < 5; ++j) { const int k = i + j - 2; if (k >= 0 && k < 256) a += c1[j] * t[n * 256 + k]; }
  u[i] = a; __syncthreads();
  float b = 0.0f;
  for (int j = 0; j < 5; ++j) { const int k = i + j - 2; if (k >= 0 && k < 256) b += c2[j] * u[k]; }
  d01[n * 256 + i] = 1.0f / (1.0f + expf(-b));
}
__global__ __launch_bounds__(256)
void fcombine_k(const float* __restrict__ xl, const float* __restrict__ xh,
                const float* __restrict__ d01, __bf16* __restrict__ fb) {
  for (int i = blockIdx.x * 256 + threadIdx.x; i < kTot; i += gridDim.x * 256) {
    const int c = (i >> 14) & 127, n = i >> 21;
    fb[i] = (__bf16)(d01[n * 256 + 2 * c] * xl[i] + d01[n * 256 + 2 * c + 1] * xh[i]);
  }
}

// ---------------------------------------------------------------- launch
extern "C" void kernel_launch(void* const* d_in, const int* in_sizes, int n_in,
                              void* d_out, int out_size, void* d_ws, size_t ws_size,
                              hipStream_t stream) {
  (void)in_sizes; (void)n_in; (void)out_size; (void)ws_size;
  const float* x        = (const float*)d_in[0];
  const float* dw_w     = (const float*)d_in[1];
  const float* dw_b     = (const float*)d_in[2];
  const float* qkvl_w   = (const float*)d_in[3];
  const float* qkvl_b   = (const float*)d_in[4];
  const float* reproj_w = (const float*)d_in[5];
  const float* reproj_b = (const float*)d_in[6];
  const float* quer_w   = (const float*)d_in[7];
  const float* quer_b   = (const float*)d_in[8];
  const float* low_w    = (const float*)d_in[9];
  const float* low_g    = (const float*)d_in[10];
  const float* low_b    = (const float*)d_in[11];
  const float* high_w   = (const float*)d_in[12];
  const float* high_g   = (const float*)d_in[13];
  const float* high_b   = (const float*)d_in[14];
  const float* c1_w     = (const float*)d_in[15];
  const float* c2_w     = (const float*)d_in[16];
  const float* oc_w     = (const float*)d_in[17];
  const float* obn_g    = (const float*)d_in[18];
  const float* obn_b    = (const float*)d_in[19];
  const float* opr_w    = (const float*)d_in[20];
  const float* opr_b    = (const float*)d_in[21];
  float* out = (float*)d_out;

  char* ws = (char*)d_ws;
  size_t off = 0;
  auto alloc = [&](size_t bytes) -> void* {
    off = (off + 255) & ~(size_t)255;
    void* p = ws + off; off += bytes; return p;
  };
  __bf16* xb    = (__bf16*)alloc((size_t)kTot * 2);
  __bf16* Bre   = (__bf16*)alloc((size_t)kTot * 2);
  __bf16* wqB   = (__bf16*)alloc(10240 * 2);
  __bf16* wreB  = (__bf16*)alloc(16384 * 2);
  __bf16* wquB  = (__bf16*)alloc(16384 * 2);
  __bf16* wprB  = (__bf16*)alloc(16384 * 2);
  __bf16* wlowB = (__bf16*)alloc(147456 * 2);
  __bf16* whiB  = (__bf16*)alloc(147456 * 2);
  __bf16* woutB = (__bf16*)alloc(147456 * 2);
  float*  qsum  = (float*)alloc((size_t)4194304 * 4);
  float*  kbuf  = (float*)alloc((size_t)4194304 * 4);
  float*  vbuf  = (float*)alloc((size_t)4194304 * 4);
  float*  qpool = (float*)alloc((size_t)1048576 * 4);
  float*  kpool = (float*)alloc((size_t)1048576 * 4);
  float*  Amat  = (float*)alloc(8192 * 4);
  __bf16* effB  = (__bf16*)alloc((size_t)kTot * 2);
  float*  qx    = (float*)alloc((size_t)kTot * 4);
  float*  sstat = (float*)alloc(2048 * 4);
  float*  ga    = (float*)alloc((size_t)kTot * 4);
  float*  gb    = (float*)alloc((size_t)kTot * 4);   // ends as ctx
  float*  cstat = (float*)alloc(2048 * 4);
  float*  att   = (float*)alloc((size_t)131072 * 4);
  __bf16* freqB = (__bf16*)alloc((size_t)kTot * 2);
  float*  xlow  = (float*)alloc((size_t)kTot * 4);
  float*  xhigh = (float*)alloc((size_t)kTot * 4);
  float*  bnpL  = (float*)alloc(256 * 4);
  float*  bnpH  = (float*)alloc(256 * 4);
  float*  bnpO  = (float*)alloc(256 * 4);
  float*  tvec  = (float*)alloc(2048 * 4);
  float*  d01   = (float*)alloc(2048 * 4);
  __bf16* fB    = (__bf16*)alloc((size_t)kTot * 2);
  __bf16* yB    = (__bf16*)alloc((size_t)kTot * 2);
  // big-buffer reuse (safe by schedule): z_low <- ga, z_high <- qx, z_out <- gb
  float* zlow = ga; float* zhigh = qx; float* zout = gb;

  const float sR = 1.2599210498948732f;         // 2^(1/3)
  const float sg0 = 1.6f, sg1 = 1.6f * sR, sg2 = 1.6f * sR * sR;

  // ---- prep: conversions + weight reorders
  cvt_bf16_k<<<8192, 256, 0, stream>>>(x, xb, kTot);
  cvt_bf16_k<<<40, 256, 0, stream>>>(qkvl_w, wqB, 10240);
  cvt_bf16_k<<<64, 256, 0, stream>>>(reproj_w, wreB, 16384);
  cvt_bf16_k<<<64, 256, 0, stream>>>(quer_w, wquB, 16384);
  cvt_bf16_k<<<64, 256, 0, stream>>>(opr_w, wprB, 16384);
  reorder3_k<<<576, 256, 0, stream>>>(low_w, wlowB);
  reorder3_k<<<576, 256, 0, stream>>>(high_w, whiB);
  reorder3_k<<<576, 256, 0, stream>>>(oc_w, woutB);
  zerof_k<<<4096, 256, 0, stream>>>(qsum, 4194304);

  // ---- branch 1: x1 / qkvl heads / pooled attention / reproj
  dwconv_gelu_k<<<512, 256, 0, stream>>>(x, dw_w, dw_b, Bre);
  qkvl_wmma_k<<<4096, 320, 0, stream>>>(xb, wqB, qkvl_b, qsum, kbuf, vbuf, Bre);
  pool_qk_k<<<256, 256, 0, stream>>>(qsum, kbuf, qpool, kpool);
  qk_softmax_k<<<8, 256, 0, stream>>>(qpool, kpool, Amat);
  x2o_k<<<512, 256, 0, stream>>>(Amat, vbuf, Bre);
  conv128_wmma_k<1, 2><<<4096, 256, 0, stream>>>(Bre, wreB, reproj_b, nullptr, effB);

  // ---- branch 2: queries / gaussian ctx / att / freq
  conv128_wmma_k<1, 1><<<4096, 256, 0, stream>>>(xb, wquB, quer_b, qx, nullptr);
  colstats_k<<<1024, 256, 0, stream>>>(qx, sstat);
  gauss_h_k<<<1024, 256, 0, stream>>>(x, ga, 3, sg0);
  gauss_v_k<<<1024, 256, 0, stream>>>(ga, gb, 3, sg0, nullptr);
  gauss_h_k<<<1024, 256, 0, stream>>>(gb, ga, 5, sg1);
  gauss_v_k<<<1024, 256, 0, stream>>>(ga, gb, 5, sg1, nullptr);
  gauss_h_k<<<1024, 256, 0, stream>>>(gb, ga, 7, sg2);
  gauss_v_k<<<1024, 256, 0, stream>>>(ga, gb, 7, sg2, x);   // ctx = x - G3  (in gb)
  rowstats_k<<<1024, 256, 0, stream>>>(gb, cstat);
  att_wmma_k<<<8, 256, 0, stream>>>(gb, cstat, att);
  freq_wmma_k<<<4096, 256, 0, stream>>>(att, qx, sstat, x, freqB);

  // ---- SFF: two 3x3 convs + BN + relu + p-norm gating
  conv128_wmma_k<9, 0><<<4096, 256, 0, stream>>>(freqB, wlowB, nullptr, zlow, nullptr);
  conv128_wmma_k<9, 0><<<4096, 256, 0, stream>>>(effB, whiB, nullptr, zhigh, nullptr);
  bnstats_k<<<128, 256, 0, stream>>>(zlow, low_g, low_b, bnpL);
  bnstats_k<<<128, 256, 0, stream>>>(zhigh, high_g, high_b, bnpH);
  bnrelu_k<0><<<16384, 256, 0, stream>>>(zlow, bnpL, xlow, nullptr);
  bnrelu_k<0><<<16384, 256, 0, stream>>>(zhigh, bnpH, xhigh, nullptr);
  pnorm_k<<<1024, 256, 0, stream>>>(xlow, tvec, 0);
  pnorm_k<<<1024, 256, 0, stream>>>(xhigh, tvec, 1);
  dvec_k<<<8, 256, 0, stream>>>(tvec, c1_w, c2_w, d01);
  fcombine_k<<<16384, 256, 0, stream>>>(xlow, xhigh, d01, fB);

  // ---- output head
  conv128_wmma_k<9, 0><<<4096, 256, 0, stream>>>(fB, woutB, nullptr, zout, nullptr);
  bnstats_k<<<128, 256, 0, stream>>>(zout, obn_g, obn_b, bnpO);
  bnrelu_k<1><<<16384, 256, 0, stream>>>(zout, bnpO, nullptr, yB);
  conv128_wmma_k<1, 1><<<4096, 256, 0, stream>>>(yB, wprB, opr_b, out, nullptr);
}
